// UNet_4836133175474
// MI455X (gfx1250) — compile-verified, matrix operand order checked by reading, not run on hardware
//
#include <hip/hip_runtime.h>
#include <hip/hip_bf16.h>

// ---------------------------------------------------------------------------
// Types / helpers for WMMA
// ---------------------------------------------------------------------------
typedef __attribute__((ext_vector_type(16))) _Float16 v16h;
typedef __attribute__((ext_vector_type(8)))  float    v8f;

#define CH     128
#define CPG    4
#define G32    32
#define NBATCH 16

union AFrag { v16h v; unsigned int u[8]; };
union HPack { _Float16 h[2]; unsigned int u; };

constexpr int ilog2(int x) { int r = 0; while (x > 1) { x >>= 1; ++r; } return r; }

// ---------------------------------------------------------------------------
// Pack conv weights f32 OIHW (Cout x Cin x 3 x 3) into f16 GEMM layout:
//   wf16[row][tap*CinPad + ci],  tap = kh*3+kw,  zero-padded in row and ci.
// CinPad is a power of two >= 32 so every 32-wide K-step has uniform tap.
// ---------------------------------------------------------------------------
__global__ __launch_bounds__(256) void pack_w_f16(
    const float* __restrict__ w, _Float16* __restrict__ out,
    int Cout, int Cin, int cinPad, int total)
{
    int idx = blockIdx.x * 256 + threadIdx.x;
    if (idx >= total) return;
    int K9  = 9 * cinPad;
    int row = idx / K9;
    int r   = idx - row * K9;
    int tap = r / cinPad;
    int ci  = r - tap * cinPad;
    float val = (row < Cout && ci < Cin)
              ? w[((size_t)row * Cin + ci) * 9 + tap] : 0.0f;
    out[idx] = (_Float16)val;
}

// ---------------------------------------------------------------------------
// Pad input x (16,3,64,64) -> (16,32,64,64) so the input conv needs no
// per-channel guards.
// ---------------------------------------------------------------------------
__global__ __launch_bounds__(256) void pad_in_kernel(
    const float* __restrict__ x, float* __restrict__ out)
{
    int idx = blockIdx.x * 256 + threadIdx.x;     // total = 16*32*4096 = 2^21
    int pix = idx & 4095;
    int c   = (idx >> 12) & 31;
    int b   = idx >> 17;
    out[idx] = (c < 3) ? x[((size_t)b * 3 + c) * 4096 + pix] : 0.0f;
}

// ---------------------------------------------------------------------------
// Implicit-GEMM 3x3 conv (pad=1) via V_WMMA_F32_16X16X32_F16.
// Templated on resolution/stride/CinPad: all index math folds to shifts,
// B gathers become immediate-offset loads off one clamped base per tap,
// tap loop fully unrolled (kh/kw literal). One wave = 16 cout x 32 pixels,
// A fragment reused across two B tiles -> 2 WMMAs per K-step.
// ---------------------------------------------------------------------------
template<int HIN, int STRIDE, int CINPAD>
__global__ __launch_bounds__(32) void conv3x3_wmma(
    const float* __restrict__ in, const unsigned int* __restrict__ wf16,
    const float* __restrict__ bias, const float* __restrict__ add0,
    const float* __restrict__ add1, float* __restrict__ out, int Cout)
{
    constexpr int WIN   = HIN;
    constexpr int HWIN  = HIN * WIN;
    constexpr int HOUT  = HIN / STRIDE;
    constexpr int WOUT  = HOUT;
    constexpr int HWOUT = HOUT * WOUT;
    constexpr int K32   = 9 * CINPAD;
    constexpr int LHW   = ilog2(HWOUT);
    constexpr int LW    = ilog2(WOUT);

    const int lane = threadIdx.x & 31;
    const int half = lane >> 4;          // 0: lanes 0-15, 1: lanes 16-31
    const int mn   = lane & 15;
    const int coutBase = blockIdx.x * 16;
    const int pixBase  = blockIdx.y * 32;

    // totalPix = 16*HWOUT is always a multiple of 32 -> no pixel guards.
    int p0 = pixBase + mn, p1 = pixBase + 16 + mn;
    int bI0 = p0 >> LHW,          bI1 = p1 >> LHW;
    int rm0 = p0 & (HWOUT - 1),   rm1 = p1 & (HWOUT - 1);
    int oy0 = rm0 >> LW,          oy1 = rm1 >> LW;
    int ox0 = rm0 & (WOUT - 1),   ox1 = rm1 & (WOUT - 1);
    int iy00 = oy0 * STRIDE - 1,  iy01 = oy1 * STRIDE - 1;
    int ix00 = ox0 * STRIDE - 1,  ix01 = ox1 * STRIDE - 1;
    const float* __restrict__ inB0 = in + (size_t)bI0 * CINPAD * HWIN;
    const float* __restrict__ inB1 = in + (size_t)bI1 * CINPAD * HWIN;

    const unsigned int* __restrict__ wRow =
        wf16 + (size_t)(coutBase + mn) * (K32 >> 1);

    v8f acc0 = {}, acc1 = {};
#pragma unroll
    for (int tap = 0; tap < 9; ++tap) {
        const int kh = tap / 3, kw = tap % 3;      // literals after unroll
        int iy0 = iy00 + kh, ix0 = ix00 + kw;
        int iy1 = iy01 + kh, ix1 = ix01 + kw;
        bool ok0 = (iy0 >= 0) & (iy0 < HIN) & (ix0 >= 0) & (ix0 < WIN);
        bool ok1 = (iy1 >= 0) & (iy1 < HIN) & (ix1 >= 0) & (ix1 < WIN);
        // Clamped per-tap base: when !ok we still read a valid in-batch
        // address and mask the result to zero.
        const float* __restrict__ base0 = ok0 ? (inB0 + iy0 * WIN + ix0) : inB0;
        const float* __restrict__ base1 = ok1 ? (inB1 + iy1 * WIN + ix1) : inB1;
#pragma unroll 2
        for (int cb = 0; cb < CINPAD; cb += 32) {
            const int k0 = tap * CINPAD + cb;
            AFrag a, b0, b1;
#pragma unroll
            for (int v = 0; v < 8; ++v) {
                // A: packed f16 pairs; dwords contiguous in v&3 -> b128 pairs.
                a.u[v] = wRow[(k0 >> 1) + ((v >> 2) << 3) + (half << 2) + (v & 3)];
                // B: slot s = K offset inside the 32-wide step (constant).
                const int s = ((v >> 2) << 4) + (half << 3) + ((v & 3) << 1);
                HPack t0, t1;
                t0.h[0] = (_Float16)base0[(size_t)(cb + s)     * HWIN];
                t0.h[1] = (_Float16)base0[(size_t)(cb + s + 1) * HWIN];
                t1.h[0] = (_Float16)base1[(size_t)(cb + s)     * HWIN];
                t1.h[1] = (_Float16)base1[(size_t)(cb + s + 1) * HWIN];
                b0.u[v] = ok0 ? t0.u : 0u;
                b1.u[v] = ok1 ? t1.u : 0u;
            }
            acc0 = __builtin_amdgcn_wmma_f32_16x16x32_f16(
                false, a.v, false, b0.v, (short)0, acc0, false, false);
            acc1 = __builtin_amdgcn_wmma_f32_16x16x32_f16(
                false, a.v, false, b1.v, (short)0, acc1, false, false);
        }
    }

    // Epilogue: D rows M = r + 8*half, col = mn.
#pragma unroll
    for (int r = 0; r < 8; ++r) {
        int co = coutBase + r + (half << 3);
        if (co >= Cout) continue;
        float bv = bias ? bias[co] : 0.0f;
        size_t o0 = ((size_t)bI0 * Cout + co) * HWOUT + rm0;
        size_t o1 = ((size_t)bI1 * Cout + co) * HWOUT + rm1;
        float val0 = acc0[r] + bv;
        float val1 = acc1[r] + bv;
        if (add0) { val0 += add0[o0]; val1 += add0[o1]; }
        if (add1) { val0 += add1[o0]; val1 += add1[o1]; }
        out[o0] = val0;
        out[o1] = val1;
    }
}

// ---------------------------------------------------------------------------
// GroupNorm (G=32, CPG=4) + per-group affine + per-(b,c) scale/shift + SiLU.
// ---------------------------------------------------------------------------
__global__ __launch_bounds__(256) void gn_affine_silu(
    const float* __restrict__ in, float* __restrict__ out,
    const float* __restrict__ gs, const float* __restrict__ gb,
    const float* __restrict__ scsh, int HW)
{
    int b = blockIdx.x >> 5;
    int g = blockIdx.x & 31;
    int n = CPG * HW;
    const float* base = in + ((size_t)b * CH + g * CPG) * HW;

    float s = 0.0f, s2 = 0.0f;
    for (int i = threadIdx.x; i < n; i += 256) {
        float x = base[i];
        s += x; s2 += x * x;
    }
    __shared__ float red0[256], red1[256];
    red0[threadIdx.x] = s; red1[threadIdx.x] = s2;
    __syncthreads();
    for (int off = 128; off > 0; off >>= 1) {
        if (threadIdx.x < off) {
            red0[threadIdx.x] += red0[threadIdx.x + off];
            red1[threadIdx.x] += red1[threadIdx.x + off];
        }
        __syncthreads();
    }
    float mean = red0[0] / (float)n;
    float var  = red1[0] / (float)n - mean * mean;
    float rstd = rsqrtf(var + 1e-8f);
    float gsc = gs[g], gsh = gb[g];

    float* ob = out + ((size_t)b * CH + g * CPG) * HW;
    for (int i = threadIdx.x; i < n; i += 256) {
        int c = g * CPG + i / HW;
        float sc = scsh[b * 2 * CH + c];
        float sh = scsh[b * 2 * CH + CH + c];
        float x = (base[i] - mean) * rstd * gsc + gsh;
        x = x * sc + sh;
        ob[i] = x / (1.0f + __expf(-x));
    }
}

// ---------------------------------------------------------------------------
// Nearest-neighbor 2x upsample (NCHW).
// ---------------------------------------------------------------------------
__global__ void up2_kernel(const float* __restrict__ in, float* __restrict__ out,
                           int C, int Hin, int Win, int total)
{
    int idx = blockIdx.x * blockDim.x + threadIdx.x;
    if (idx >= total) return;
    int Wout = Win * 2, Hout = Hin * 2;
    int x = idx % Wout; int t = idx / Wout;
    int y = t % Hout;  t /= Hout;
    int c = t % C;     int b = t / C;
    out[idx] = in[(((size_t)b * C + c) * Hin + (y >> 1)) * Win + (x >> 1)];
}

// ---------------------------------------------------------------------------
// Bottleneck attention at hw=4, ch=128 (replicates reference reshape quirk).
// ---------------------------------------------------------------------------
__global__ __launch_bounds__(128) void attn_kernel(
    const float* __restrict__ in, float* __restrict__ out,
    const float* __restrict__ wq, const float* __restrict__ bq,
    const float* __restrict__ wk, const float* __restrict__ bk,
    const float* __restrict__ wv, const float* __restrict__ bv,
    const float* __restrict__ wo, const float* __restrict__ bo)
{
    int b = blockIdx.x;
    int c = threadIdx.x;
    __shared__ float xf[4][128], q[4][128], k[4][128], v[4][128], A[4][4], o[4][128];
    const float* xb = in + (size_t)b * 512;
    for (int i = 0; i < 4; ++i) xf[i][c] = xb[i * 128 + c];
    __syncthreads();
    for (int i = 0; i < 4; ++i) {
        float sq = bq[c], sk = bk[c], sv = bv[c];
        for (int d = 0; d < 128; ++d) {
            float xv = xf[i][d];
            sq += xv * wq[c * 128 + d];
            sk += xv * wk[c * 128 + d];
            sv += xv * wv[c * 128 + d];
        }
        q[i][c] = sq; k[i][c] = sk; v[i][c] = sv;
    }
    __syncthreads();
    if (c < 16) {
        int i = c >> 2, j = c & 3;
        float s = 0.0f;
        for (int d = 0; d < 128; ++d) s += q[i][d] * k[j][d];
        A[i][j] = s / sqrtf(128.0f);
    }
    __syncthreads();
    if (c < 4) {
        float m = fmaxf(fmaxf(A[c][0], A[c][1]), fmaxf(A[c][2], A[c][3]));
        float e0 = __expf(A[c][0] - m), e1 = __expf(A[c][1] - m);
        float e2 = __expf(A[c][2] - m), e3 = __expf(A[c][3] - m);
        float inv = 1.0f / (e0 + e1 + e2 + e3);
        A[c][0] = e0 * inv; A[c][1] = e1 * inv; A[c][2] = e2 * inv; A[c][3] = e3 * inv;
    }
    __syncthreads();
    for (int i = 0; i < 4; ++i)
        o[i][c] = A[i][0] * v[0][c] + A[i][1] * v[1][c] + A[i][2] * v[2][c] + A[i][3] * v[3][c];
    __syncthreads();
    float* ob = out + (size_t)b * 512;
    for (int i = 0; i < 4; ++i) {
        float s = bo[c];
        for (int d = 0; d < 128; ++d) s += o[i][d] * wo[c * 128 + d];
        ob[c * 4 + i] = s;
    }
}

// ---------------------------------------------------------------------------
// Time embedding: sinusoid -> lin(64->256) -> SiLU -> lin(256->64).
// ---------------------------------------------------------------------------
__global__ __launch_bounds__(256) void time_embed_kernel(
    const float* __restrict__ t, const float* __restrict__ w1,
    const float* __restrict__ b1, const float* __restrict__ w2,
    const float* __restrict__ b2, float* __restrict__ te)
{
    __shared__ float e[NBATCH][64];
    __shared__ float h1[NBATCH][256];
    int tid = threadIdx.x;
    for (int idx = tid; idx < NBATCH * 64; idx += 256) {
        int b = idx >> 6, j = idx & 63;
        int i = j & 31;
        float freq = __expf(-(float)i * logf(10000.0f) / 32.0f);
        float arg = t[b] * freq;
        e[b][j] = (j < 32) ? cosf(arg) : sinf(arg);
    }
    __syncthreads();
    for (int idx = tid; idx < NBATCH * 256; idx += 256) {
        int b = idx >> 8, o = idx & 255;
        float s = b1[o];
        for (int d = 0; d < 64; ++d) s += e[b][d] * w1[o * 64 + d];
        h1[b][o] = s / (1.0f + __expf(-s));
    }
    __syncthreads();
    for (int idx = tid; idx < NBATCH * 64; idx += 256) {
        int b = idx >> 6, o = idx & 63;
        float s = b2[o];
        for (int d = 0; d < 256; ++d) s += h1[b][d] * w2[o * 256 + d];
        te[idx] = s;
    }
}

// ---------------------------------------------------------------------------
// scale/shift: scsh[b, 0:256] = te[b] @ ss_w.T + ss_b
// ---------------------------------------------------------------------------
__global__ __launch_bounds__(256) void scsh_kernel(
    const float* __restrict__ te, const float* __restrict__ ssw,
    const float* __restrict__ ssb, float* __restrict__ scsh)
{
    int idx = blockIdx.x * 256 + threadIdx.x;
    int b = idx >> 8, o = idx & 255;
    float s = ssb[o];
    for (int d = 0; d < 64; ++d) s += te[b * 64 + d] * ssw[o * 64 + d];
    scsh[idx] = s;
}

// ---------------------------------------------------------------------------
// Host orchestration
// ---------------------------------------------------------------------------
struct PackedW { const unsigned int* ptr; };

template<int HIN, int STRIDE, int CINPAD>
static void launchConv(const float* in, PackedW pw, const float* bias,
                       const float* a0, const float* a1, float* out,
                       int Cout, hipStream_t stream)
{
    constexpr int HOUT = HIN / STRIDE;
    dim3 grid((Cout + 15) / 16, (NBATCH * HOUT * HOUT + 31) / 32);
    conv3x3_wmma<HIN, STRIDE, CINPAD><<<grid, 32, 0, stream>>>(
        in, pw.ptr, bias, a0, a1, out, Cout);
}

extern "C" void kernel_launch(void* const* d_in, const int* in_sizes, int n_in,
                              void* d_out, int out_size, void* d_ws, size_t ws_size,
                              hipStream_t stream)
{
    (void)in_sizes; (void)n_in; (void)out_size; (void)ws_size;
    auto F = [&](int i) { return (const float*)d_in[i]; };

    const float* x    = F(0);
    const float* tvec = F(1);
    const int DBASE = 8, MBASE = 48, UBASE = 62, SSW = 92, SSB = 103;

    // Workspace layout (floats)
    float* ws = (float*)d_ws;
    float* te   = ws;                 // 16*64
    float* scsh = ws + 1024;          // 16*256
    const size_t WREGION = 2 * 1024 * 1024;   // floats: packed f16 weights (8 MB)
    _Float16* wbase = (_Float16*)(ws + 8192);
    size_t whalf = 0;
    const size_t XPAD = 2 * 1024 * 1024;      // 16*32*4096 floats
    float* xpad = ws + 8192 + WREGION;
    const size_t A64 = (size_t)NBATCH * CH * 64 * 64;
    size_t off = 8192 + WREGION + XPAD;
    float* B0 = ws + off; off += A64;
    float* B1 = ws + off; off += A64;
    float* B2 = ws + off; off += A64;
    float* B3 = ws + off; off += A64;
    float* R[5];
    R[0] = ws + off; off += A64;
    R[1] = ws + off; off += A64 / 4;
    R[2] = ws + off; off += A64 / 16;
    R[3] = ws + off; off += A64 / 64;
    R[4] = ws + off; off += A64 / 256;

    auto prep = [&](const float* w, int Cout, int Cin, int cinPad) -> PackedW {
        int K32 = 9 * cinPad;
        int CoutPad = (Cout + 15) & ~15;
        _Float16* dst = wbase + whalf;
        int total = CoutPad * K32;
        whalf += (size_t)total;
        pack_w_f16<<<(total + 255) / 256, 256, 0, stream>>>(w, dst, Cout, Cin, cinPad, total);
        return PackedW{(const unsigned int*)dst};
    };

    // 128-channel conv dispatcher (templated on res/stride)
    auto conv = [&](const float* in, PackedW pw, const float* bias,
                    const float* a0, const float* a1, float* out,
                    int Cout, int Hin, int stride) {
        if (stride == 1) {
            switch (Hin) {
            case 64: launchConv<64, 1, 128>(in, pw, bias, a0, a1, out, Cout, stream); break;
            case 32: launchConv<32, 1, 128>(in, pw, bias, a0, a1, out, Cout, stream); break;
            case 16: launchConv<16, 1, 128>(in, pw, bias, a0, a1, out, Cout, stream); break;
            case  8: launchConv< 8, 1, 128>(in, pw, bias, a0, a1, out, Cout, stream); break;
            case  4: launchConv< 4, 1, 128>(in, pw, bias, a0, a1, out, Cout, stream); break;
            case  2: launchConv< 2, 1, 128>(in, pw, bias, a0, a1, out, Cout, stream); break;
            }
        } else {
            switch (Hin) {
            case 64: launchConv<64, 2, 128>(in, pw, bias, a0, a1, out, Cout, stream); break;
            case 32: launchConv<32, 2, 128>(in, pw, bias, a0, a1, out, Cout, stream); break;
            case 16: launchConv<16, 2, 128>(in, pw, bias, a0, a1, out, Cout, stream); break;
            case  8: launchConv< 8, 2, 128>(in, pw, bias, a0, a1, out, Cout, stream); break;
            case  4: launchConv< 4, 2, 128>(in, pw, bias, a0, a1, out, Cout, stream); break;
            }
        }
    };
    auto gn = [&](const float* in, float* out, const float* gs, const float* gb, int res) {
        gn_affine_silu<<<NBATCH * G32, 256, 0, stream>>>(in, out, gs, gb, scsh, res * res);
    };
    auto up2 = [&](const float* in, float* out, int res) {
        int total = NBATCH * CH * 4 * res * res;
        up2_kernel<<<(total + 255) / 256, 256, 0, stream>>>(in, out, CH, res, res, total);
    };
    auto ss = [&](int idx) {
        scsh_kernel<<<16, 256, 0, stream>>>(te, F(SSW + idx), F(SSB + idx), scsh);
    };

    // --- pack all conv weights to f16 (tap-major, CinPad-padded) ---
    PackedW PWin  = prep(F(2), CH, 3, 32);
    PackedW PWout = prep(F(3), 3, CH, 128);
    PackedW Dc1[5], Dc2[5], Dsk[5];
    for (int i = 0; i < 5; ++i) {
        Dc1[i] = prep(F(DBASE + i * 8 + 4), CH, CH, 128);
        Dc2[i] = prep(F(DBASE + i * 8 + 5), CH, CH, 128);
        Dsk[i] = prep(F(DBASE + i * 8 + 6), CH, CH, 128);
    }
    PackedW Mc1 = prep(F(MBASE + 4), CH, CH, 128);
    PackedW Mc2 = prep(F(MBASE + 5), CH, CH, 128);
    PackedW Uc1[5], Uc2[5];
    for (int i = 0; i < 5; ++i) {
        Uc1[i] = prep(F(UBASE + i * 6 + 4), CH, CH, 128);
        Uc2[i] = prep(F(UBASE + i * 6 + 5), CH, CH, 128);
    }

    // --- time embedding + padded input ---
    time_embed_kernel<<<1, 256, 0, stream>>>(tvec, F(4), F(5), F(6), F(7), te);
    pad_in_kernel<<<(16 * 32 * 4096) / 256, 256, 0, stream>>>(x, xpad);

    // --- input conv: xpad(16,32,64,64) -> R[0] ---
    launchConv<64, 1, 32>(xpad, PWin, nullptr, nullptr, nullptr, R[0], CH, stream);

    // --- down blocks ---
    int res = 64;
    float* cur = R[0];
    for (int i = 0; i < 5; ++i) {
        const float* gn1s = F(DBASE + i * 8 + 0); const float* gn1b = F(DBASE + i * 8 + 1);
        const float* gn2s = F(DBASE + i * 8 + 2); const float* gn2b = F(DBASE + i * 8 + 3);
        const float* skb  = F(DBASE + i * 8 + 7);
        float* outBuf = (i < 4) ? R[i + 1] : B0;
        ss(i);
        gn(cur, B1, gn1s, gn1b, res);
        conv(B1, Dc1[i], nullptr, nullptr, nullptr, B2, CH, res, 2);
        gn(B2, B1, gn2s, gn2b, res / 2);
        conv(cur, Dsk[i], skb, nullptr, nullptr, B3, CH, res, 2);      // og
        conv(B1, Dc2[i], nullptr, B3, nullptr, outBuf, CH, res / 2, 1);
        cur = outBuf;
        res >>= 1;
    }

    // --- bottleneck (res = 2) ---
    {
        const float* gn1s = F(MBASE + 0); const float* gn1b = F(MBASE + 1);
        const float* gn2s = F(MBASE + 2); const float* gn2b = F(MBASE + 3);
        const float* wq = F(MBASE + 6), *wk = F(MBASE + 7), *wv = F(MBASE + 8), *wo = F(MBASE + 9);
        const float* bq = F(MBASE + 10), *bk = F(MBASE + 11), *bv = F(MBASE + 12), *bo = F(MBASE + 13);
        ss(5);
        gn(cur, B1, gn1s, gn1b, res);
        attn_kernel<<<NBATCH, 128, 0, stream>>>(B1, B2, wq, bq, wk, bk, wv, bv, wo, bo);
        conv(B2, Mc1, nullptr, nullptr, nullptr, B1, CH, res, 1);
        gn(B1, B2, gn2s, gn2b, res);
        attn_kernel<<<NBATCH, 128, 0, stream>>>(B2, B3, wq, bq, wk, bk, wv, bv, wo, bo);
        conv(B3, Mc2, nullptr, cur, nullptr, B1, CH, res, 1);
        cur = B1;
    }

    // --- up blocks ---
    for (int i = 0; i < 5; ++i) {
        const float* gn1s = F(UBASE + i * 6 + 0); const float* gn1b = F(UBASE + i * 6 + 1);
        const float* gn2s = F(UBASE + i * 6 + 2); const float* gn2b = F(UBASE + i * 6 + 3);
        float* t1 = B0, *t2 = B2, *t3 = B3;       // cur lives in B1 throughout
        ss(6 + i);
        up2(cur, t1, res);                                    // og = up2(x)
        gn(cur, t2, gn1s, gn1b, res);
        up2(t2, t3, res);
        conv(t3, Uc1[i], nullptr, nullptr, nullptr, t2, CH, res * 2, 1);
        gn(t2, t3, gn2s, gn2b, res * 2);
        conv(t3, Uc2[i], nullptr, t1, R[4 - i], cur, CH, res * 2, 1);
        res <<= 1;
    }

    // --- output conv: cur(16,128,64,64) -> d_out(16,3,64,64) ---
    conv(cur, PWout, nullptr, nullptr, nullptr, (float*)d_out, 3, 64, 1);
}
